// EncoderMixtureModelTrajectory_56487409877141
// MI455X (gfx1250) — matrix operand order repaired and checked by reading, not compiled, not run on hardware
//
#include <hip/hip_runtime.h>

typedef __attribute__((ext_vector_type(16))) _Float16 v16h;
typedef __attribute__((ext_vector_type(8)))  _Float16 v8h;
typedef __attribute__((ext_vector_type(8)))  float    v8f;
typedef __attribute__((ext_vector_type(4)))  float    vf4;

#define DIM    512
#define BM     64          // batch rows per workgroup
#define LDSW   520         // padded LDS row stride (halfs); 1040B = 16B-aligned rows
#define KSTEPS (DIM / 32)  // 16 wmma K-steps per tile
#define BATCH  65536
#define NCLS   16
#define NLAT   16

// ---------------- weight packing: f32 -> f16 (row-major kept) ----------------
__global__ void pack_f16(const float* __restrict__ src, _Float16* __restrict__ dst, int n) {
  int i = blockIdx.x * 256 + threadIdx.x;
  if (i < n) dst[i] = (_Float16)src[i];
}

// Wg is (16 classes, 32, 512); we only need e<16 (mu). dst row n = c*16+e.
__global__ void pack_wg_mu(const float* __restrict__ Wg, _Float16* __restrict__ dst) {
  int i = blockIdx.x * 256 + threadIdx.x;
  if (i >= NCLS * NLAT * DIM) return;
  int n = i >> 9, k = i & (DIM - 1);
  int c = n >> 4, e = n & 15;
  dst[i] = (_Float16)Wg[((size_t)c * 32 + e) * DIM + k];
}

// --------------- one 16x16 output tile, K=512 accumulated via WMMA -----------
// ap   : LDS pointer, pre-offset to this lane's A row/quadrant
// wrow : global f16 weights, pre-offset to this lane's B column/K-chunk
__device__ __forceinline__ v8f tile_dot(const _Float16* __restrict__ ap,
                                        const _Float16* __restrict__ wrow) {
  v8f acc = {};
#pragma unroll
  for (int k = 0; k < KSTEPS; ++k) {
    const int kb = k * 32;
    // A 16-bit layout: lanes 0-15 hold K {kb..kb+7, kb+16..kb+23}, lanes 16-31
    // hold K {kb+8..kb+15, kb+24..kb+31}; ap already includes (lane>>4)*8.
    v8h alo = *(const v8h*)(ap + kb);
    v8h ahi = *(const v8h*)(ap + kb + 16);
    v16h a = __builtin_shufflevector(alo, ahi, 0,1,2,3,4,5,6,7,8,9,10,11,12,13,14,15);
    // B layout: lane L = column N=L&15, K {kb..kb+7, kb+16..kb+23} offset by (L>>4)*8.
    v8h blo = *(const v8h*)(wrow + kb);
    v8h bhi = *(const v8h*)(wrow + kb + 16);
    v16h b = __builtin_shufflevector(blo, bhi, 0,1,2,3,4,5,6,7,8,9,10,11,12,13,14,15);
    acc = __builtin_amdgcn_wmma_f32_16x16x32_f16(false, a, false, b, (short)0, acc,
                                                 false, false);
  }
  return acc;
}

// --------------- dense layer: out = act(in @ W^T + b), all in LDS ------------
__device__ __forceinline__ void layer_pass(const _Float16* __restrict__ in,
                                           _Float16* __restrict__ out,
                                           const _Float16* __restrict__ W,
                                           const float* __restrict__ bias,
                                           bool relu, int rg, int nhalf, int lhi,
                                           int llo) {
  const int arow = rg * 16 + llo;
  const _Float16* ap = in + arow * LDSW + lhi * 8;
  for (int t = 0; t < 16; ++t) {          // 16 N-tiles of 16 per wave (N-half)
    const int N = nhalf * 256 + t * 16 + llo;
    const _Float16* wrow = W + (size_t)N * DIM + lhi * 8;
    v8f acc = tile_dot(ap, wrow);
    const float bval = bias[N];
#pragma unroll
    for (int r = 0; r < 8; ++r) {         // C/D layout: M = r + (lane>>4)*8
      int M = rg * 16 + r + lhi * 8;
      float v = acc[r] + bval;
      if (relu) v = v > 0.f ? v : 0.f;
      out[M * LDSW + N] = (_Float16)v;
    }
  }
}

// ------------------------------- fused kernel --------------------------------
__global__ __launch_bounds__(256) void fused_mlp(
    const float* __restrict__ x,
    const _Float16* __restrict__ W0f, const float* __restrict__ b0,
    const _Float16* __restrict__ W1f, const float* __restrict__ b1,
    const _Float16* __restrict__ W2f, const float* __restrict__ b2,
    const _Float16* __restrict__ Wcf, const float* __restrict__ bc,
    const _Float16* __restrict__ Wgf, const float* __restrict__ bg,
    float* __restrict__ zout, int* __restrict__ yout) {
  __shared__ _Float16 sbuf0[BM * LDSW];
  __shared__ _Float16 sbuf1[BM * LDSW];
  const int tid = threadIdx.x;
  const int lane = tid & 31, wave = tid >> 5;
  const int lhi = lane >> 4, llo = lane & 15;
  const int rg = wave & 3, nhalf = wave >> 2;  // row-group / N-half per wave
  const int row0 = blockIdx.x * BM;

  // stage x tile (f32 global) -> f16 LDS
  for (int i = tid; i < BM * DIM / 4; i += 256) {
    int r = i >> 7, c4 = i & 127;
    vf4 v = *(const vf4*)(x + (size_t)(row0 + r) * DIM + c4 * 4);
    _Float16* p = &sbuf0[r * LDSW + c4 * 4];
    p[0] = (_Float16)v.x; p[1] = (_Float16)v.y;
    p[2] = (_Float16)v.z; p[3] = (_Float16)v.w;
  }
  __syncthreads();

  layer_pass(sbuf0, sbuf1, W0f, b0, true,  rg, nhalf, lhi, llo);
  __syncthreads();
  layer_pass(sbuf1, sbuf0, W1f, b1, true,  rg, nhalf, lhi, llo);
  __syncthreads();
  layer_pass(sbuf0, sbuf1, W2f, b2, false, rg, nhalf, lhi, llo);
  __syncthreads();

  // m lives in sbuf1; logits (f32) + all-class mu (f16, bg folded) -> sbuf0
  const int arow = rg * 16 + llo;
  const _Float16* ap = sbuf1 + arow * LDSW + lhi * 8;

  if (nhalf == 0) {  // classifier: one 16-wide tile per row-group
    const _Float16* wrow = Wcf + (size_t)llo * DIM + lhi * 8;
    v8f acc = tile_dot(ap, wrow);
    const float bval = bc[llo];
#pragma unroll
    for (int r = 0; r < 8; ++r) {
      int M = rg * 16 + r + lhi * 8;
      float* lp = (float*)(sbuf0 + M * LDSW + 256);  // byte off 512, 16B aligned
      lp[llo] = acc[r] + bval;
    }
  }
  for (int t = 0; t < 8; ++t) {  // mu: 256 columns split across N-halves
    int N = nhalf * 128 + t * 16 + llo;
    const _Float16* wrow = Wgf + (size_t)N * DIM + lhi * 8;
    v8f acc = tile_dot(ap, wrow);
    const float bval = bg[(N >> 4) * 32 + (N & 15)];  // bg[class][e], e<16
#pragma unroll
    for (int r = 0; r < 8; ++r) {
      int M = rg * 16 + r + lhi * 8;
      sbuf0[M * LDSW + N] = (_Float16)(acc[r] + bval);
    }
  }
  __syncthreads();

  // argmax + gather + writeback (softmax is monotone -> argmax of logits)
  if (tid < BM) {
    const float* lp = (const float*)(sbuf0 + tid * LDSW + 256);
    int best = 0;
    float bv = lp[0];
#pragma unroll
    for (int c = 1; c < NCLS; ++c) {
      float v = lp[c];
      if (v > bv) { bv = v; best = c; }
    }
    const _Float16* mz = sbuf0 + tid * LDSW + best * 16;
    size_t g = (size_t)(row0 + tid);
#pragma unroll
    for (int e = 0; e < NLAT; ++e) zout[g * NLAT + e] = (float)mz[e];
    yout[g] = best;
  }
}

// ---------------------------------- launcher ---------------------------------
extern "C" void kernel_launch(void* const* d_in, const int* in_sizes, int n_in,
                              void* d_out, int out_size, void* d_ws, size_t ws_size,
                              hipStream_t stream) {
  const float* x  = (const float*)d_in[0];
  const float* W0 = (const float*)d_in[1];
  const float* b0 = (const float*)d_in[2];
  const float* W1 = (const float*)d_in[3];
  const float* b1 = (const float*)d_in[4];
  const float* W2 = (const float*)d_in[5];
  const float* b2 = (const float*)d_in[6];
  const float* Wc = (const float*)d_in[7];
  const float* bc = (const float*)d_in[8];
  const float* Wg = (const float*)d_in[9];
  const float* bg = (const float*)d_in[10];

  _Float16* W0f = (_Float16*)d_ws;
  _Float16* W1f = W0f + (size_t)DIM * DIM;
  _Float16* W2f = W1f + (size_t)DIM * DIM;
  _Float16* Wcf = W2f + (size_t)DIM * DIM;
  _Float16* Wgf = Wcf + (size_t)NCLS * DIM;

  const int nw = DIM * DIM;
  pack_f16<<<(nw + 255) / 256, 256, 0, stream>>>(W0, W0f, nw);
  pack_f16<<<(nw + 255) / 256, 256, 0, stream>>>(W1, W1f, nw);
  pack_f16<<<(nw + 255) / 256, 256, 0, stream>>>(W2, W2f, nw);
  pack_f16<<<(NCLS * DIM + 255) / 256, 256, 0, stream>>>(Wc, Wcf, NCLS * DIM);
  pack_wg_mu<<<(NCLS * NLAT * DIM + 255) / 256, 256, 0, stream>>>(Wg, Wgf);

  float* zout = (float*)d_out;
  int* yout = (int*)d_out + (size_t)BATCH * NLAT;
  fused_mlp<<<BATCH / BM, 256, 0, stream>>>(x, W0f, b0, W1f, b1, W2f, b2, Wcf, bc,
                                            Wgf, bg, zout, yout);
}
